// Attention_18846316494887
// MI455X (gfx1250) — compile-verified, matrix operand order checked by reading, not run on hardware
//
#include <hip/hip_runtime.h>
#include <hip/hip_bf16.h>

// ---------------------------------------------------------------------------
// Problem constants (match reference)
// ---------------------------------------------------------------------------
#define E_DIM 1024
#define H_NUM 16
#define D_DIM 64
#define R_DIM 8
#define BATCH 2
#define SEQ   2048
#define TOK   (BATCH * SEQ)      // 4096 tokens
#define NQKV  (3 * E_DIM)        // 3072

typedef __attribute__((ext_vector_type(16))) _Float16 v16h;
typedef __attribute__((ext_vector_type(8)))  _Float16 v8h;
typedef __attribute__((ext_vector_type(8)))  float    v8f;

__device__ __forceinline__ v8f wmma16(v16h a, v16h b, v8f c) {
  // D = A(16x32 f16) * B(32x16 f16) + C(16x16 f32)
  return __builtin_amdgcn_wmma_f32_16x16x32_f16(
      /*neg_a=*/false, a, /*neg_b=*/false, b,
      /*c_mod=*/(short)0, c, /*reuse_a=*/false, /*reuse_b=*/false);
}

// Load one 16x32 f16 operand tile into the wave32 WMMA register layout.
// Element (r, k) comes from base[(major + r) * ld + kb + k].
//   lane%16 -> r (A: row M / B: col N), lane/16 selects K-halves:
//   halves 0..7  = K (lane/16)*8 + 0..7
//   halves 8..15 = K 16 + (lane/16)*8 + 0..7
__device__ __forceinline__ v16h load_tile16(const _Float16* base, int ld,
                                            int major, int kb, int lane) {
  const _Float16* p = base + (size_t)(major + (lane & 15)) * (size_t)ld
                           + (size_t)(kb + ((lane >> 4) << 3));
  v8h lo = *(const v8h*)p;        // K = koff + 0..7
  v8h hi = *(const v8h*)(p + 16); // K = koff + 16..23
  return __builtin_shufflevector(lo, hi, 0,1,2,3,4,5,6,7,8,9,10,11,12,13,14,15);
}

// ---------------------------------------------------------------------------
// 0) f32 -> f16 convert
// ---------------------------------------------------------------------------
__global__ void cvt_f32_to_f16(const float* __restrict__ in,
                               _Float16* __restrict__ out, int n) {
  int i = blockIdx.x * blockDim.x + threadIdx.x;
  if (i < n) out[i] = (_Float16)in[i];
}

// ---------------------------------------------------------------------------
// 1) U = y @ [Bq | Bk | Bv]   -> [TOK, 24] f32
// ---------------------------------------------------------------------------
__global__ __launch_bounds__(256)
void lora_u_kernel(const float* __restrict__ y, const float* __restrict__ Bq,
                   const float* __restrict__ Bk, const float* __restrict__ Bv,
                   float* __restrict__ U) {
  int t = blockIdx.x, tid = threadIdx.x;
  const float* yr = y + (size_t)t * E_DIM;
  float acc[24];
#pragma unroll
  for (int j = 0; j < 24; ++j) acc[j] = 0.f;
  for (int e = tid; e < E_DIM; e += 256) {
    float yv = yr[e];
#pragma unroll
    for (int r = 0; r < R_DIM; ++r) {
      acc[r]      += yv * Bq[e * R_DIM + r];
      acc[8 + r]  += yv * Bk[e * R_DIM + r];
      acc[16 + r] += yv * Bv[e * R_DIM + r];
    }
  }
  __shared__ float red[256];
  for (int j = 0; j < 24; ++j) {
    red[tid] = acc[j];
    __syncthreads();
    for (int s = 128; s > 0; s >>= 1) {
      if (tid < s) red[tid] += red[tid + s];
      __syncthreads();
    }
    if (tid == 0) U[(size_t)t * 24 + j] = red[0];
    __syncthreads();
  }
}

// ---------------------------------------------------------------------------
// 2) QKV GEMM: C[TOK, 3072] = Xh @ Wqkv^T + bias + LoRA, scatter to
//    Qh,Kh [B,H,S,D] f16 (Q pre-scaled by 1/sqrt(D)) and Vt [B,H,D,S] f16.
//    Wave tile: 32(M) x 64(N), 8 WMMAs per 32-k step (A reused 4x, B reused 2x).
//    Block: 4 waves = 32 x 256.
// ---------------------------------------------------------------------------
__global__ __launch_bounds__(128)
void qkv_wmma_kernel(const _Float16* __restrict__ Xh,
                     const _Float16* __restrict__ Wh,     // [3072,1024] f16
                     const float* __restrict__ bqkv,
                     const float* __restrict__ U,         // [TOK,24]
                     const float* __restrict__ Aq, const float* __restrict__ Ak,
                     const float* __restrict__ Av,
                     _Float16* __restrict__ Qh, _Float16* __restrict__ Kh,
                     _Float16* __restrict__ Vt) {
  int lane = threadIdx.x & 31, wave = threadIdx.x >> 5;
  int rowb = blockIdx.y * 32;
  int colb = blockIdx.x * 256 + wave * 64;

  v8f acc[2][4];
#pragma unroll
  for (int i = 0; i < 2; ++i)
#pragma unroll
    for (int j = 0; j < 4; ++j) acc[i][j] = v8f{};

  for (int kb = 0; kb < E_DIM; kb += 32) {
    v16h a0 = load_tile16(Xh, E_DIM, rowb, kb, lane);
    v16h a1 = load_tile16(Xh, E_DIM, rowb + 16, kb, lane);
    __builtin_prefetch(Xh + (size_t)(rowb + (lane & 15)) * E_DIM + kb + 64, 0, 1);
#pragma unroll
    for (int j = 0; j < 4; ++j) {
      v16h b = load_tile16(Wh, E_DIM, colb + j * 16, kb, lane);
      acc[0][j] = wmma16(a0, b, acc[0][j]);
      acc[1][j] = wmma16(a1, b, acc[1][j]);
    }
  }

#pragma unroll
  for (int j = 0; j < 4; ++j) {
    int col = colb + j * 16 + (lane & 15);   // 0..3071
    int g = col >> 10;                       // 0=Q 1=K 2=V (uniform per tile)
    int e = col & 1023;
    const float* Ag = (g == 0) ? Aq : ((g == 1) ? Ak : Av);
    float a8[R_DIM];
#pragma unroll
    for (int r = 0; r < R_DIM; ++r) a8[r] = Ag[r * E_DIM + e];
    float bias = bqkv[col];
    int hh = e >> 6, d = e & 63;
#pragma unroll
    for (int i = 0; i < 2; ++i) {
#pragma unroll
      for (int v = 0; v < 8; ++v) {
        int row = rowb + i * 16 + v + ((lane >> 4) << 3);  // token index
        float val = acc[i][j][v] + bias;
        const float* Ur = U + (size_t)row * 24 + g * R_DIM;
#pragma unroll
        for (int r = 0; r < R_DIM; ++r) val += Ur[r] * a8[r];
        int bb = row >> 11, ss = row & (SEQ - 1);
        if (g == 0) {
          Qh[(((size_t)bb * H_NUM + hh) * SEQ + ss) * D_DIM + d] =
              (_Float16)(val * 0.125f);               // fold 1/sqrt(64)
        } else if (g == 1) {
          Kh[(((size_t)bb * H_NUM + hh) * SEQ + ss) * D_DIM + d] = (_Float16)val;
        } else {
          Vt[(((size_t)bb * H_NUM + hh) * D_DIM + d) * SEQ + ss] = (_Float16)val;
        }
      }
    }
  }
}

// ---------------------------------------------------------------------------
// 3) Flash attention: one wave per 32-query tile, online softmax,
//    16 WMMAs per 32-key step (V operand reused across both query subtiles).
//    Output Oh [TOK, E] f16 in [B,S,H*D] order.
// ---------------------------------------------------------------------------
__global__ __launch_bounds__(256)
void attn_wmma_kernel(const _Float16* __restrict__ Q,
                      const _Float16* __restrict__ K,
                      const _Float16* __restrict__ Vt,
                      _Float16* __restrict__ Oh) {
  int lane = threadIdx.x & 31, wave = threadIdx.x >> 5;
  int id = blockIdx.x * 8 + wave;           // 0 .. B*H*(S/32)-1 = 2047
  int qt = id & (SEQ / 32 - 1);             // 32-query tile in sequence
  int bh = id >> 6;                         // 0..31
  int b = bh >> 4, h = bh & 15;
  const _Float16* Qp = Q + (size_t)bh * SEQ * D_DIM;
  const _Float16* Kp = K + (size_t)bh * SEQ * D_DIM;
  const _Float16* Vp = Vt + (size_t)bh * D_DIM * SEQ;
  int qrow = qt * 32;

  v16h aq[2][2];
#pragma unroll
  for (int i = 0; i < 2; ++i) {
    aq[i][0] = load_tile16(Qp, D_DIM, qrow + i * 16, 0, lane);
    aq[i][1] = load_tile16(Qp, D_DIM, qrow + i * 16, 32, lane);
  }

  v8f o[2][4];
  float m[2][8], l[2][8];
#pragma unroll
  for (int i = 0; i < 2; ++i) {
#pragma unroll
    for (int j = 0; j < 4; ++j) o[i][j] = v8f{};
#pragma unroll
    for (int v = 0; v < 8; ++v) { m[i][v] = -1e30f; l[i][v] = 0.f; }
  }

  // per-wave staging: P is 32x32 f16, stride 40 halves (80B rows, 16B-aligned)
  __shared__ __align__(16) _Float16 pb[8][32 * 40];
  _Float16* pw = &pb[wave][0];

  for (int kb = 0; kb < SEQ; kb += 32) {
    v8f s[2][2];
#pragma unroll
    for (int i = 0; i < 2; ++i) { s[i][0] = v8f{}; s[i][1] = v8f{}; }

#pragma unroll
    for (int kt = 0; kt < 2; ++kt) {
      v16h bk0 = load_tile16(Kp, D_DIM, kb + kt * 16, 0, lane);
      v16h bk1 = load_tile16(Kp, D_DIM, kb + kt * 16, 32, lane);
#pragma unroll
      for (int i = 0; i < 2; ++i) {
        s[i][kt] = wmma16(aq[i][0], bk0, s[i][kt]);
        s[i][kt] = wmma16(aq[i][1], bk1, s[i][kt]);
      }
    }

#pragma unroll
    for (int i = 0; i < 2; ++i) {
#pragma unroll
      for (int v = 0; v < 8; ++v) {
        float x0 = s[i][0][v], x1 = s[i][1][v];
        float mx = fmaxf(x0, x1);
#pragma unroll
        for (int off = 1; off < 16; off <<= 1)
          mx = fmaxf(mx, __shfl_xor(mx, off, 32));  // stays in 16-lane half
        float mn = fmaxf(m[i][v], mx);
        float p0 = __expf(x0 - mn);
        float p1 = __expf(x1 - mn);
        float rs = p0 + p1;
#pragma unroll
        for (int off = 1; off < 16; off <<= 1) rs += __shfl_xor(rs, off, 32);
        float corr = __expf(m[i][v] - mn);
        l[i][v] = l[i][v] * corr + rs;
        m[i][v] = mn;
        o[i][0][v] *= corr; o[i][1][v] *= corr;
        o[i][2][v] *= corr; o[i][3][v] *= corr;
        int row = i * 16 + v + ((lane >> 4) << 3);
        int col = lane & 15;
        pw[row * 40 + col]      = (_Float16)p0;     // keys kb..kb+15
        pw[row * 40 + 16 + col] = (_Float16)p1;     // keys kb+16..kb+31
      }
    }

    // V operand tiles loaded once, reused by both query subtiles
    v16h bv[4];
#pragma unroll
    for (int j = 0; j < 4; ++j) bv[j] = load_tile16(Vp, SEQ, j * 16, kb, lane);
    // re-lane P (C layout) into A layout via LDS (DS ops in-order per wave)
#pragma unroll
    for (int i = 0; i < 2; ++i) {
      v16h ap = load_tile16(pw + i * 16 * 40, 40, 0, 0, lane);
#pragma unroll
      for (int j = 0; j < 4; ++j) o[i][j] = wmma16(ap, bv[j], o[i][j]);
    }
  }

#pragma unroll
  for (int i = 0; i < 2; ++i) {
#pragma unroll
    for (int v = 0; v < 8; ++v) {
      float inv = 1.0f / l[i][v];
      int ss = qrow + i * 16 + v + ((lane >> 4) << 3);
      size_t rowoff = ((size_t)b * SEQ + ss) * E_DIM;
      int colb = h * D_DIM + (lane & 15);
      Oh[rowoff + colb +  0] = (_Float16)(o[i][0][v] * inv);
      Oh[rowoff + colb + 16] = (_Float16)(o[i][1][v] * inv);
      Oh[rowoff + colb + 32] = (_Float16)(o[i][2][v] * inv);
      Oh[rowoff + colb + 48] = (_Float16)(o[i][3][v] * inv);
    }
  }
}

// ---------------------------------------------------------------------------
// 4) Uo = O @ Bo  -> [TOK, 8] f32
// ---------------------------------------------------------------------------
__global__ __launch_bounds__(256)
void lora_uo_kernel(const _Float16* __restrict__ Oh,
                    const float* __restrict__ Bo, float* __restrict__ Uo) {
  int t = blockIdx.x, tid = threadIdx.x;
  const _Float16* orow = Oh + (size_t)t * E_DIM;
  float acc[R_DIM];
#pragma unroll
  for (int r = 0; r < R_DIM; ++r) acc[r] = 0.f;
  for (int e = tid; e < E_DIM; e += 256) {
    float ov = (float)orow[e];
#pragma unroll
    for (int r = 0; r < R_DIM; ++r) acc[r] += ov * Bo[e * R_DIM + r];
  }
  __shared__ float red[256];
  for (int j = 0; j < R_DIM; ++j) {
    red[tid] = acc[j];
    __syncthreads();
    for (int s = 128; s > 0; s >>= 1) {
      if (tid < s) red[tid] += red[tid + s];
      __syncthreads();
    }
    if (tid == 0) Uo[(size_t)t * R_DIM + j] = red[0];
    __syncthreads();
  }
}

// ---------------------------------------------------------------------------
// 5) res = O @ Wmsa^T + O @ (Bo Ao) + y   (f32 out)
//    Wave tile 32x64 like the QKV GEMM.
// ---------------------------------------------------------------------------
__global__ __launch_bounds__(128)
void msa_wmma_kernel(const _Float16* __restrict__ Oh,
                     const _Float16* __restrict__ Wmh,   // [1024,1024] f16
                     const float* __restrict__ Uo, const float* __restrict__ Ao,
                     const float* __restrict__ y, float* __restrict__ res) {
  int lane = threadIdx.x & 31, wave = threadIdx.x >> 5;
  int rowb = blockIdx.y * 32;
  int colb = blockIdx.x * 256 + wave * 64;

  v8f acc[2][4];
#pragma unroll
  for (int i = 0; i < 2; ++i)
#pragma unroll
    for (int j = 0; j < 4; ++j) acc[i][j] = v8f{};

  for (int kb = 0; kb < E_DIM; kb += 32) {
    v16h a0 = load_tile16(Oh, E_DIM, rowb, kb, lane);
    v16h a1 = load_tile16(Oh, E_DIM, rowb + 16, kb, lane);
    __builtin_prefetch(Oh + (size_t)(rowb + (lane & 15)) * E_DIM + kb + 64, 0, 1);
#pragma unroll
    for (int j = 0; j < 4; ++j) {
      v16h b = load_tile16(Wmh, E_DIM, colb + j * 16, kb, lane);
      acc[0][j] = wmma16(a0, b, acc[0][j]);
      acc[1][j] = wmma16(a1, b, acc[1][j]);
    }
  }

#pragma unroll
  for (int j = 0; j < 4; ++j) {
    int col = colb + j * 16 + (lane & 15);
    float a8[R_DIM];
#pragma unroll
    for (int r = 0; r < R_DIM; ++r) a8[r] = Ao[r * E_DIM + col];
#pragma unroll
    for (int i = 0; i < 2; ++i) {
#pragma unroll
      for (int v = 0; v < 8; ++v) {
        int row = rowb + i * 16 + v + ((lane >> 4) << 3);
        float val = acc[i][j][v];
        const float* Ur = Uo + (size_t)row * R_DIM;
#pragma unroll
        for (int r = 0; r < R_DIM; ++r) val += Ur[r] * a8[r];
        val += y[(size_t)row * E_DIM + col];
        res[(size_t)row * E_DIM + col] = val;
      }
    }
  }
}

// ---------------------------------------------------------------------------
// 6) LayerNorm rows of res -> d_out
// ---------------------------------------------------------------------------
__global__ __launch_bounds__(256)
void layernorm_kernel(const float* __restrict__ res,
                      const float* __restrict__ gamma,
                      const float* __restrict__ beta, float* __restrict__ out) {
  int t = blockIdx.x, tid = threadIdx.x;
  const float* rr = res + (size_t)t * E_DIM;
  float s1 = 0.f, s2 = 0.f;
  for (int e = tid; e < E_DIM; e += 256) {
    float x = rr[e];
    s1 += x; s2 += x * x;
  }
  __shared__ float r1[256], r2[256];
  r1[tid] = s1; r2[tid] = s2;
  __syncthreads();
  for (int s = 128; s > 0; s >>= 1) {
    if (tid < s) { r1[tid] += r1[tid + s]; r2[tid] += r2[tid + s]; }
    __syncthreads();
  }
  float mu = r1[0] * (1.0f / E_DIM);
  float var = r2[0] * (1.0f / E_DIM) - mu * mu;
  float sc = rsqrtf(var + 1e-6f);
  for (int e = tid; e < E_DIM; e += 256)
    out[(size_t)t * E_DIM + e] = (rr[e] - mu) * sc * gamma[e] + beta[e];
}

// ---------------------------------------------------------------------------
// Host launcher
// ---------------------------------------------------------------------------
extern "C" void kernel_launch(void* const* d_in, const int* in_sizes, int n_in,
                              void* d_out, int out_size, void* d_ws, size_t ws_size,
                              hipStream_t stream) {
  const float* y    = (const float*)d_in[0];
  const float* Wqkv = (const float*)d_in[1];
  const float* bqkv = (const float*)d_in[2];
  const float* Wmsa = (const float*)d_in[3];
  const float* Bq   = (const float*)d_in[4];
  const float* Aq   = (const float*)d_in[5];
  const float* Bk   = (const float*)d_in[6];
  const float* Ak   = (const float*)d_in[7];
  const float* Bv   = (const float*)d_in[8];
  const float* Av   = (const float*)d_in[9];
  const float* Bo   = (const float*)d_in[10];
  const float* Ao   = (const float*)d_in[11];
  const float* gamma = (const float*)d_in[12];
  const float* beta  = (const float*)d_in[13];
  float* out = (float*)d_out;

  // workspace carve-up (all offsets 256B-aligned)
  char* ws = (char*)d_ws;
  size_t off = 0;
  auto carve = [&](size_t bytes) {
    char* p = ws + off;
    off += (bytes + 255) & ~(size_t)255;
    return p;
  };
  _Float16* Xh   = (_Float16*)carve((size_t)TOK * E_DIM * 2);   // y f16
  _Float16* Wqh  = (_Float16*)carve((size_t)NQKV * E_DIM * 2);  // Wqkv f16
  _Float16* Wmh  = (_Float16*)carve((size_t)E_DIM * E_DIM * 2); // Wmsa f16
  float*    U    = (float*)carve((size_t)TOK * 24 * 4);
  _Float16* Qh   = (_Float16*)carve((size_t)TOK * E_DIM * 2);   // [B,H,S,D]
  _Float16* Kh   = (_Float16*)carve((size_t)TOK * E_DIM * 2);   // [B,H,S,D]
  _Float16* Vt   = (_Float16*)carve((size_t)TOK * E_DIM * 2);   // [B,H,D,S]
  _Float16* Oh   = (_Float16*)carve((size_t)TOK * E_DIM * 2);   // [B,S,E]
  float*    Uo   = (float*)carve((size_t)TOK * R_DIM * 4);
  float*    res  = (float*)carve((size_t)TOK * E_DIM * 4);
  (void)ws_size; (void)in_sizes; (void)n_in; (void)out_size;

  // 0) converts
  {
    int n = TOK * E_DIM;
    cvt_f32_to_f16<<<(n + 255) / 256, 256, 0, stream>>>(y, Xh, n);
    n = NQKV * E_DIM;
    cvt_f32_to_f16<<<(n + 255) / 256, 256, 0, stream>>>(Wqkv, Wqh, n);
    n = E_DIM * E_DIM;
    cvt_f32_to_f16<<<(n + 255) / 256, 256, 0, stream>>>(Wmsa, Wmh, n);
  }
  // 1) LoRA U
  lora_u_kernel<<<TOK, 256, 0, stream>>>(y, Bq, Bk, Bv, U);
  // 2) QKV GEMM (grid: N/256 x M/32)
  qkv_wmma_kernel<<<dim3(NQKV / 256, TOK / 32), 128, 0, stream>>>(
      Xh, Wqh, bqkv, U, Aq, Ak, Av, Qh, Kh, Vt);
  // 3) attention: B*H*(S/32) wave tiles, 8 waves/block
  attn_wmma_kernel<<<(BATCH * H_NUM * (SEQ / 32)) / 8, 256, 0, stream>>>(
      Qh, Kh, Vt, Oh);
  // 4) LoRA Uo
  lora_uo_kernel<<<TOK, 256, 0, stream>>>(Oh, Bo, Uo);
  // 5) output projection + residual (grid: N/256 x M/32)
  msa_wmma_kernel<<<dim3(E_DIM / 256, TOK / 32), 128, 0, stream>>>(
      Oh, Wmh, Uo, Ao, y, res);
  // 6) LayerNorm
  layernorm_kernel<<<TOK, 256, 0, stream>>>(res, gamma, beta, out);
}
